// PWBLinearLayer_11673721110974
// MI455X (gfx1250) — matrix operand order repaired and checked by reading, not compiled, not run on hardware
//
#include <hip/hip_runtime.h>

typedef __attribute__((ext_vector_type(16))) _Float16 v16h;
typedef __attribute__((ext_vector_type(8)))  _Float16 v8h;
typedef __attribute__((ext_vector_type(4)))  _Float16 v4h;
typedef __attribute__((ext_vector_type(8)))  float    v8f;
typedef int gv4i __attribute__((vector_size(16)));   // matches async-LDS builtin param type

#define AS1 __attribute__((address_space(1)))
#define AS3 __attribute__((address_space(3)))

// Problem dims (fixed by the reference harness)
constexpr int BATCH = 4096;
constexpr int DIN   = 2048;
constexpr int DOUT  = 2048;

// GEMM tiling: 128x256 block tile, 8 waves of 64x64, BK=64, double-buffered LDS
constexpr int BM  = 128;
constexpr int BN  = 256;
constexpr int BK  = 64;
constexpr int LDT = BK + 8;     // LDS row stride in halves: 144 B = 36 dwords -> conflict-free b128
constexpr int NT  = DIN / BK;   // 32 K-tiles

#if defined(__has_builtin)
#if __has_builtin(__builtin_amdgcn_global_load_async_to_lds_b128) && \
    __has_builtin(__builtin_amdgcn_s_wait_asynccnt)
#define USE_ASYNC_LDS 1
#endif
#endif
#ifndef USE_ASYNC_LDS
#define USE_ASYNC_LDS 0
#endif

__device__ __forceinline__ float pwb_q(float v) {
    v = fminf(fmaxf(v, -1.0f), 1.0f);
    return rintf(v * 127.0f) / 127.0f;   // RNE matches jnp.round
}

// ---- x: f32 -> f16 straight conversion (x is NOT quantized in the reference) ----
__global__ __launch_bounds__(256) void cvt_x_kernel(const float* __restrict__ x,
                                                    _Float16* __restrict__ xh) {
    int i = (blockIdx.x * 256 + threadIdx.x) * 4;
    float4 v = *(const float4*)(x + i);
    v4h o;
    o[0] = (_Float16)v.x; o[1] = (_Float16)v.y;
    o[2] = (_Float16)v.z; o[3] = (_Float16)v.w;
    *(v4h*)(xh + i) = o;
}

// ---- W: quantize + transpose into Wt[n][k] f16 so B-fragments are K-contiguous ----
__global__ __launch_bounds__(256) void quant_wt_kernel(const float* __restrict__ W,
                                                       _Float16* __restrict__ wt) {
    __shared__ float tile[32][33];
    const int n0 = blockIdx.x * 32;
    const int k0 = blockIdx.y * 32;
    const int tx = threadIdx.x;
    #pragma unroll
    for (int j = threadIdx.y; j < 32; j += 8)
        tile[j][tx] = W[(size_t)(k0 + j) * DOUT + n0 + tx];
    __syncthreads();
    #pragma unroll
    for (int j = threadIdx.y; j < 32; j += 8) {
        float v = pwb_q(tile[tx][j]);            // = q(W[k0+tx][n0+j])
        wt[(size_t)(n0 + j) * DIN + k0 + tx] = (_Float16)v;
    }
}

// ---- b: quantize (kept in f32 for the epilogue) ----
__global__ __launch_bounds__(256) void quant_b_kernel(const float* __restrict__ b,
                                                      float* __restrict__ bq) {
    int i = blockIdx.x * 256 + threadIdx.x;
    bq[i] = pwb_q(b[i]);
}

// ---- main WMMA GEMM: out = relu(xh @ Wt^T + bq) ----
__global__ __launch_bounds__(256)
__attribute__((amdgpu_waves_per_eu(1, 8)))
void pwb_gemm_kernel(const _Float16* __restrict__ xh,
                     const _Float16* __restrict__ wt,
                     const float* __restrict__ bq,
                     float* __restrict__ out) {
#if USE_ASYNC_LDS
    __shared__ __align__(16) _Float16 As[2][BM * LDT];   // 2 x 18 KB
    __shared__ __align__(16) _Float16 Bs[2][BN * LDT];   // 2 x 36 KB
#else
    __shared__ __align__(16) _Float16 As[1][BM * LDT];
    __shared__ __align__(16) _Float16 Bs[1][BN * LDT];
#endif

    const int tid  = threadIdx.x;
    const int lane = tid & 31;
    const int wid  = tid >> 5;          // 8 waves
    const int wm   = wid & 1;           // 2 waves in M  -> 64 rows each
    const int wn   = wid >> 1;          // 4 waves in N  -> 64 cols each
    const int l16  = lane & 15;
    const int lhi  = lane >> 4;         // half-wave select

    const int bm = blockIdx.x;          // 0..31
    const int bn = blockIdx.y;          // 0..7

    // cooperative tile loader mapping: b128 per thread covers 8 halves of K
    const int r0 = tid >> 3;            // 0..31, rows r0 + 32*i
    const int kc = tid & 7;             // 8-half chunk within BK

    const _Float16* aglb = xh + (size_t)(bm * BM) * DIN;
    const _Float16* bglb = wt + (size_t)(bn * BN) * DIN;

    v8f acc[4][4] = {};                 // 4 M-tiles x 4 N-tiles per wave (64x64)

#if USE_ASYNC_LDS
    // Issue 12 async global->LDS b128 copies (4 for A's 128 rows, 8 for B's 256 rows).
    auto issue_tile = [&](int kt, int buf) {
        #pragma unroll
        for (int i = 0; i < 4; ++i) {
            const int r = r0 + 32 * i;
            __builtin_amdgcn_global_load_async_to_lds_b128(
                (AS1 gv4i*)(aglb + (size_t)r * DIN + kt * BK + kc * 8),
                (AS3 gv4i*)&As[buf][r * LDT + kc * 8], 0, 0);
        }
        #pragma unroll
        for (int i = 0; i < 8; ++i) {
            const int r = r0 + 32 * i;
            __builtin_amdgcn_global_load_async_to_lds_b128(
                (AS1 gv4i*)(bglb + (size_t)r * DIN + kt * BK + kc * 8),
                (AS3 gv4i*)&Bs[buf][r * LDT + kc * 8], 0, 0);
        }
    };
    issue_tile(0, 0);
#else
    float4 areg[4], breg[8];
    auto load_tile = [&](int kt) {
        #pragma unroll
        for (int i = 0; i < 4; ++i)
            areg[i] = *(const float4*)(aglb + (size_t)(r0 + 32 * i) * DIN + kt * BK + kc * 8);
        #pragma unroll
        for (int i = 0; i < 8; ++i)
            breg[i] = *(const float4*)(bglb + (size_t)(r0 + 32 * i) * DIN + kt * BK + kc * 8);
    };
#endif

    #pragma unroll 1
    for (int kt = 0; kt < NT; ++kt) {
#if USE_ASYNC_LDS
        const int buf = kt & 1;
        if (kt + 1 < NT) {
            issue_tile(kt + 1, (kt + 1) & 1);      // overlap next tile's DMA with this compute
            __builtin_amdgcn_s_wait_asynccnt(12);  // this wave's 12 copies for tile kt complete
        } else {
            __builtin_amdgcn_s_wait_asynccnt(0);
        }
        __syncthreads();                           // all waves' copies for tile kt visible
#else
        const int buf = 0;
        load_tile(kt);
        #pragma unroll
        for (int i = 0; i < 4; ++i)
            *(float4*)&As[0][(r0 + 32 * i) * LDT + kc * 8] = areg[i];
        #pragma unroll
        for (int i = 0; i < 8; ++i)
            *(float4*)&Bs[0][(r0 + 32 * i) * LDT + kc * 8] = breg[i];
        __syncthreads();
#endif

        #pragma unroll
        for (int kk = 0; kk < BK; kk += 32) {
            v16h afrag[4], bfrag[4];
            // A 16x32 f16 layout: lanes<16 hold K{0..7,16..23}, lanes>=16 K{8..15,24..31}
            #pragma unroll
            for (int im = 0; im < 4; ++im) {
                const _Float16* p = &As[buf][(wm * 64 + im * 16 + l16) * LDT + kk + lhi * 8];
                v8h lo = *(const v8h*)p;
                v8h hi = *(const v8h*)(p + 16);
                afrag[im] = __builtin_shufflevector(lo, hi,
                    0,1,2,3,4,5,6,7,8,9,10,11,12,13,14,15);
            }
            // B 32x16 f16 layout: lanes<16 hold K{0..7,16..23}, lanes>=16 K{8..15,24..31}
            // of column (lane&15), matching A
            #pragma unroll
            for (int in = 0; in < 4; ++in) {
                const _Float16* p = &Bs[buf][(wn * 64 + in * 16 + l16) * LDT + kk + lhi * 8];
                v8h lo = *(const v8h*)p;
                v8h hi = *(const v8h*)(p + 16);
                bfrag[in] = __builtin_shufflevector(lo, hi,
                    0,1,2,3,4,5,6,7,8,9,10,11,12,13,14,15);
            }
            #pragma unroll
            for (int im = 0; im < 4; ++im)
                #pragma unroll
                for (int in = 0; in < 4; ++in)
                    acc[im][in] = __builtin_amdgcn_wmma_f32_16x16x32_f16(
                        false, afrag[im], false, bfrag[in],
                        (short)0, acc[im][in], false, false);
        }
        __syncthreads();
    }

    // Epilogue: bias + ReLU.  C/D layout: VGPR g -> M = g (+8 for lanes>=16), lane&15 -> N
    #pragma unroll
    for (int im = 0; im < 4; ++im) {
        const int grow = bm * BM + wm * 64 + im * 16 + lhi * 8;
        #pragma unroll
        for (int in = 0; in < 4; ++in) {
            const int gcol = bn * BN + wn * 64 + in * 16 + l16;
            const float bias = bq[gcol];
            #pragma unroll
            for (int g = 0; g < 8; ++g) {
                float v = acc[im][in][g] + bias;
                out[(size_t)(grow + g) * DOUT + gcol] = fmaxf(v, 0.0f);
            }
        }
    }
}

extern "C" void kernel_launch(void* const* d_in, const int* in_sizes, int n_in,
                              void* d_out, int out_size, void* d_ws, size_t ws_size,
                              hipStream_t stream) {
    const float* x = (const float*)d_in[0];
    const float* W = (const float*)d_in[1];
    const float* b = (const float*)d_in[2];
    float* out = (float*)d_out;

    // workspace layout (~25.2 MB): xh | Wt | bq
    char* ws = (char*)d_ws;
    _Float16* xh = (_Float16*)ws;                                   // 4096*2048*2 B
    _Float16* wt = (_Float16*)(ws + (size_t)BATCH * DIN * 2);       // 2048*2048*2 B
    float*    bq = (float*)(ws + (size_t)BATCH * DIN * 2
                               + (size_t)DIN * DOUT * 2);           // 2048*4 B

    cvt_x_kernel<<<dim3((BATCH * DIN) / (4 * 256)), 256, 0, stream>>>(x, xh);
    quant_wt_kernel<<<dim3(DOUT / 32, DIN / 32), dim3(32, 8), 0, stream>>>(W, wt);
    quant_b_kernel<<<dim3(DOUT / 256), 256, 0, stream>>>(b, bq);

    pwb_gemm_kernel<<<dim3(BATCH / BM, DOUT / BN), 256, 0, stream>>>(xh, wt, bq, out);
}
